// FlashMoELayer_77146202570781
// MI455X (gfx1250) — compile-verified, matrix operand order checked by reading, not run on hardware
//
#include <hip/hip_runtime.h>
#include <hip/hip_bf16.h>

#define T_TOK   4096       // B*S
#define HDIM    1024
#define EEXP    32
#define DDIM    1024
#define KCHUNK  64
#define MTILE   32         // tokens per block tile (2 WMMA M-tiles)

typedef __attribute__((ext_vector_type(8)))  float  v8f;
typedef __attribute__((ext_vector_type(2)))  __bf16 v2bf;
typedef __attribute__((ext_vector_type(8)))  __bf16 v8bf;
typedef __attribute__((ext_vector_type(16))) __bf16 v16bf;

// Packed f32x2 -> bf16x2.
//  - native v_cvt_pk_bf16_f32 if the toolchain declares it (1 op/pair)
//  - else: bits+0x8000 per float (round-half-up at the bf16 boundary) and a
//    single v_perm_b32 grabbing both high halves (3 VALU ops/pair total).
__device__ __forceinline__ unsigned int pack_bf16x2(float a, float b) {
#if __has_builtin(__builtin_amdgcn_cvt_pk_bf16_f32)
    v2bf v = __builtin_amdgcn_cvt_pk_bf16_f32(a, b);
    return __builtin_bit_cast(unsigned int, v);
#else
    unsigned int ua = __float_as_uint(a) + 0x8000u;
    unsigned int ub = __float_as_uint(b) + 0x8000u;
    // result bytes: [1:0] = ua[3:2], [3:2] = ub[3:2]  (sel 4-7 -> src0, 0-3 -> src1)
    return __builtin_amdgcn_perm(ub, ua, 0x07060302u);
#endif
}

// ---------------------------------------------------------------------------
// Kernel 0: zero output (float4) + counters
// ---------------------------------------------------------------------------
__global__ __launch_bounds__(256) void moe_zero_kernel(float4* out4, int n4, int* cnt) {
    int i = blockIdx.x * blockDim.x + threadIdx.x;
    if (i < n4) out4[i] = make_float4(0.f, 0.f, 0.f, 0.f);
    if (blockIdx.x == 0 && threadIdx.x < EEXP) cnt[threadIdx.x] = 0;
}

// ---------------------------------------------------------------------------
// Kernel 1: router (softmax + top-2) + dispatch append.
// One wave per token; lane e owns expert e (E == 32 == wave32).
// ---------------------------------------------------------------------------
__global__ __launch_bounds__(256) void moe_router_kernel(
    const float* __restrict__ x, const float* __restrict__ rw,
    int* __restrict__ cnt, int* __restrict__ tok, float* __restrict__ wgt) {

    __shared__ float xs[8][HDIM];
    const int tid  = threadIdx.x;
    const int lane = tid & 31;
    const int wv   = tid >> 5;
    const int t    = blockIdx.x * 8 + wv;          // grid = T/8, always valid

    const float* xp = x + (size_t)t * HDIM;
    for (int i = lane; i < HDIM; i += 32) xs[wv][i] = xp[i];
    __syncthreads();

    float acc = 0.f;
    const float4* r4 = (const float4*)(rw + (size_t)lane * HDIM);
    const float4* x4 = (const float4*)(&xs[wv][0]);
    #pragma unroll 8
    for (int j = 0; j < HDIM / 4; ++j) {
        float4 a = x4[j], b = r4[j];
        acc += a.x * b.x + a.y * b.y + a.z * b.z + a.w * b.w;
    }

    // Softmax over 32 lanes
    float m = acc;
    for (int off = 16; off; off >>= 1) m = fmaxf(m, __shfl_xor(m, off, 32));
    float p = __expf(acc - m);
    float s = p;
    for (int off = 16; off; off >>= 1) s += __shfl_xor(s, off, 32);
    p /= s;

    // Top-1 (value, index) butterfly reduction
    float v1 = p; int i1 = lane;
    for (int off = 16; off; off >>= 1) {
        float ov = __shfl_xor(v1, off, 32);
        int   oi = __shfl_xor(i1, off, 32);
        if (ov > v1 || (ov == v1 && oi < i1)) { v1 = ov; i1 = oi; }
    }
    // Top-2: mask out winner
    float v2 = (lane == i1) ? -1.f : p; int i2 = lane;
    for (int off = 16; off; off >>= 1) {
        float ov = __shfl_xor(v2, off, 32);
        int   oi = __shfl_xor(i2, off, 32);
        if (ov > v2 || (ov == v2 && oi < i2)) { v2 = ov; i2 = oi; }
    }

    if (lane == 0) {
        int pos = atomicAdd(&cnt[i1], 1);
        tok[i1 * T_TOK + pos] = t; wgt[i1 * T_TOK + pos] = v1;
        pos = atomicAdd(&cnt[i2], 1);
        tok[i2 * T_TOK + pos] = t; wgt[i2 * T_TOK + pos] = v2;
    }
}

// ---------------------------------------------------------------------------
// Kernel 2: grouped expert GEMM via bf16 WMMA, double-buffered LDS pipeline.
// grid = (D/128, T/32, E).  block = 256 threads = 8 waves.
// Block tile: 32 gathered tokens x 128 cols; each wave holds TWO 16x16 f32
// accumulators sharing one B fragment (2x L2 reuse of the weight panel).
// Pipeline: global loads (chunk i+1) -> WMMA (chunk i, buf p) -> cvt+ds_store
// (chunk i+1, buf p^1) -> one barrier.
// ---------------------------------------------------------------------------
__global__ __launch_bounds__(256) void moe_gemm_kernel(
    const float* __restrict__ x, const float* __restrict__ ew,
    const int* __restrict__ cnt, const int* __restrict__ tok,
    const float* __restrict__ wgt, float* __restrict__ out) {

    const int e     = blockIdx.z;
    const int count = cnt[e];
    const int tile  = blockIdx.y;
    if (tile * MTILE >= count) return;              // block-uniform exit
    const int nb = blockIdx.x * 128;                // output-column base

    __shared__ int            s_tid[MTILE];
    __shared__ float          s_tw[MTILE];
    __shared__ unsigned short sA[2][MTILE * KCHUNK];   // [row][k]  bf16
    __shared__ unsigned short sB[2][128 * KCHUNK];     // [n][k]    bf16 (K-transposed)

    const int tid = threadIdx.x;

    if (tid < MTILE) {
        int r = tile * MTILE + tid;
        if (r < count) { s_tid[tid] = tok[e * T_TOK + r]; s_tw[tid] = wgt[e * T_TOK + r]; }
        else           { s_tid[tid] = -1;                 s_tw[tid] = 0.f; }
    }
    __syncthreads();

    const float* W = ew + (size_t)e * HDIM * DDIM;  // [K=H][N=D] row-major

    const int lane = tid & 31;
    const int wv   = tid >> 5;                      // wave id: 0..7 -> 16-col slab
    const int mr   = lane & 15;                     // A row / B col within tile
    const int kb   = (lane < 16) ? 0 : 8;           // per-lane K base (ISA layout)

    // A staging: 32 rows x 16 quads of 4 K-values -> 2 quads per thread
    const int arow = tid >> 4;                      // 0..15 (also handles arow+16)
    const int akq  = tid & 15;
    // B staging: 128 cols x 2 K-phases
    const int bn  = tid & 127;
    const int bkp = tid >> 7;

    // ---- staging registers ----
    float4 av[2];
    float  bw[32];

    // Global loads for one K-chunk into registers
    auto load_chunk = [&](int kk) {
        #pragma unroll
        for (int j = 0; j < 2; ++j) {
            int trow = s_tid[arow + 16 * j];
            av[j] = make_float4(0.f, 0.f, 0.f, 0.f);
            if (trow >= 0)
                av[j] = ((const float4*)(x + (size_t)trow * HDIM + kk))[akq];
        }
        const float* Wb = W + (size_t)kk * DDIM + nb + bn;
        #pragma unroll
        for (int rr = 0; rr < 16; ++rr) {
            int k = 2 * bkp + 4 * rr;               // even k in [0,64)
            bw[2 * rr]     = Wb[(size_t)k * DDIM];
            bw[2 * rr + 1] = Wb[(size_t)(k + 1) * DDIM];
        }
    };
    // Convert (add + v_perm_b32) + store registers into LDS buffer p
    auto store_chunk = [&](int p) {
        #pragma unroll
        for (int j = 0; j < 2; ++j) {
            unsigned int lo = pack_bf16x2(av[j].x, av[j].y);
            unsigned int hi = pack_bf16x2(av[j].z, av[j].w);
            *(uint2*)&sA[p][(arow + 16 * j) * KCHUNK + 4 * akq] = make_uint2(lo, hi);
        }
        #pragma unroll
        for (int rr = 0; rr < 16; ++rr) {
            int k = 2 * bkp + 4 * rr;
            *(unsigned int*)&sB[p][bn * KCHUNK + k] = pack_bf16x2(bw[2 * rr], bw[2 * rr + 1]);
        }
    };

    v8f acc0 = {};
    v8f acc1 = {};

    // ---- pipeline prologue: chunk 0 ----
    load_chunk(0);
    store_chunk(0);
    __syncthreads();

    int p = 0;
    for (int kk = 0; kk < HDIM; kk += KCHUNK) {
        const bool more = (kk + KCHUNK < HDIM);
        if (more) {
            load_chunk(kk + KCHUNK);                // global loads in flight
            __builtin_prefetch(W + (size_t)(kk + 2 * KCHUNK) * DDIM + nb + bn, 0, 1);
        }

        // ---- 4 WMMAs on buffer p: two M-tiles share each B fragment ----
        #pragma unroll
        for (int sub = 0; sub < 2; ++sub) {
            const int k0 = 32 * sub;
            v8bf blo = *(const v8bf*)&sB[p][(16 * wv + mr) * KCHUNK + k0 + kb];
            v8bf bhi = *(const v8bf*)&sB[p][(16 * wv + mr) * KCHUNK + k0 + kb + 16];
            v16bf b = __builtin_shufflevector(blo, bhi,
                        0,1,2,3,4,5,6,7,8,9,10,11,12,13,14,15);

            v8bf a0lo = *(const v8bf*)&sA[p][mr * KCHUNK + k0 + kb];
            v8bf a0hi = *(const v8bf*)&sA[p][mr * KCHUNK + k0 + kb + 16];
            v16bf a0 = __builtin_shufflevector(a0lo, a0hi,
                        0,1,2,3,4,5,6,7,8,9,10,11,12,13,14,15);
            acc0 = __builtin_amdgcn_wmma_f32_16x16x32_bf16(
                       false, a0, false, b, (short)0, acc0, false, false);

            v8bf a1lo = *(const v8bf*)&sA[p][(16 + mr) * KCHUNK + k0 + kb];
            v8bf a1hi = *(const v8bf*)&sA[p][(16 + mr) * KCHUNK + k0 + kb + 16];
            v16bf a1 = __builtin_shufflevector(a1lo, a1hi,
                        0,1,2,3,4,5,6,7,8,9,10,11,12,13,14,15);
            acc1 = __builtin_amdgcn_wmma_f32_16x16x32_bf16(
                       false, a1, false, b, (short)0, acc1, false, false);
        }

        if (more) {
            store_chunk(p ^ 1);                     // fill other buffer
            __syncthreads();                        // one barrier per chunk
            p ^= 1;
        }
    }

    // ---- scaled scatter: out[token, col] += w * acc ----
    const int mbase = (lane >> 4) * 8;              // lanes 0-15 -> M 0..7, 16-31 -> M 8..15
    const int col   = nb + 16 * wv + mr;
    #pragma unroll
    for (int r = 0; r < 8; ++r) {
        int m0 = mbase + r;
        int t0 = s_tid[m0];
        if (t0 >= 0)
            atomicAdd(&out[(size_t)t0 * DDIM + col], acc0[r] * s_tw[m0]);
        int m1 = 16 + mbase + r;
        int t1 = s_tid[m1];
        if (t1 >= 0)
            atomicAdd(&out[(size_t)t1 * DDIM + col], acc1[r] * s_tw[m1]);
    }
}

// ---------------------------------------------------------------------------
extern "C" void kernel_launch(void* const* d_in, const int* in_sizes, int n_in,
                              void* d_out, int out_size, void* d_ws, size_t ws_size,
                              hipStream_t stream) {
    const float* x        = (const float*)d_in[0];   // [4,1024,1024]
    const float* router_w = (const float*)d_in[1];   // [32,1024]
    const float* ew       = (const float*)d_in[2];   // [32,1024,1024]
    float*       out      = (float*)d_out;           // [4,1024,1024]

    int*   cnt = (int*)d_ws;
    int*   tok = cnt + EEXP;
    float* wgt = (float*)(tok + EEXP * T_TOK);

    // 1) zero output + counters
    {
        int n4 = (T_TOK * DDIM) / 4;
        moe_zero_kernel<<<(n4 + 255) / 256, 256, 0, stream>>>((float4*)out, n4, cnt);
    }
    // 2) router + dispatch (one wave per token)
    moe_router_kernel<<<T_TOK / 8, 256, 0, stream>>>(x, router_w, cnt, tok, wgt);
    // 3) grouped expert GEMM (bf16 WMMA, double-buffered), scatter-accumulate
    {
        dim3 grid(DDIM / 128, T_TOK / MTILE, EEXP);  // (8, 128, 32); empty tiles exit
        moe_gemm_kernel<<<grid, 256, 0, stream>>>(x, ew, cnt, tok, wgt, out);
    }
}
